// Compressor_11192684773920
// MI455X (gfx1250) — matrix operand order repaired and checked by reading, not compiled, not run on hardware
//
#include <hip/hip_runtime.h>
#include <hip/hip_bf16.h>

// ---------------------------------------------------------------------------
// Problem constants (from the reference)
// ---------------------------------------------------------------------------
#define T_TOK  32768            // prefill tokens
#define DIM    4096             // hidden size (K of the GEMM)
#define HD     256              // compressed head dim
#define NGRP   (T_TOK / 4)      // compressed positions (RATIO = 4)
#define POOL   65536            // KV pool slots
#define ROPE_D 64

typedef __bf16 bf16_t;
typedef __attribute__((ext_vector_type(16))) __bf16 v16bf;
typedef __attribute__((ext_vector_type(8)))  float  v8f;

union Frag32B { uint4 u[2]; v16bf v; };

// ---------------------------------------------------------------------------
// Async global->LDS staging (CDNA5 ASYNCcnt path) via inline asm on gfx1250.
// The clang builtin's pointer params use HIP language address spaces that are
// not spellable from source, so we emit GLOBAL_LOAD_ASYNC_TO_LDS_B128
// directly. The LDS-address operand is the wave-relative LDS byte offset ==
// low 32 bits of the generic pointer derived from a __shared__ object.
// ---------------------------------------------------------------------------
__device__ __forceinline__ void g2l_b128(const bf16_t* g, bf16_t* l) {
#if defined(__gfx1250__)
    asm volatile("global_load_async_to_lds_b128 %0, %1, off"
                 :
                 : "v"((unsigned int)(unsigned long long)l), "v"(g)
                 : "memory");
#else
    *(uint4*)l = *(const uint4*)g;
#endif
}

template <int N>
__device__ __forceinline__ void wait_async() {
#if defined(__gfx1250__)
# if defined(__has_builtin) && __has_builtin(__builtin_amdgcn_s_wait_asynccnt)
    __builtin_amdgcn_s_wait_asynccnt(N);
# else
    asm volatile("s_wait_asynccnt %0" :: "i"(N) : "memory");
# endif
#endif
}

// ---------------------------------------------------------------------------
// Kernel 0: copy the (poisoned) output pool from the input kv_buffer
// ---------------------------------------------------------------------------
__global__ __launch_bounds__(256) void pool_copy_kernel(const float4* __restrict__ src,
                                                        float4* __restrict__ dst) {
    size_t i = (size_t)blockIdx.x * 256 + threadIdx.x;
    dst[i] = src[i];
}

// ---------------------------------------------------------------------------
// Kernel 1: fused bf16 WMMA GEMM + sigmoid gating + overlapping-window
//           compression partials.
//
// Block tile: 128 tokens x 32 "d" columns, covering all four N strips of
// kv_score mapping to those d columns (W rows s*256 + d0 + [0,32)):
//   strip 0: kv c=0 | strip 1: kv c=1 | strip 2: gate c=0 | strip 3: gate c=1
//
// 8 waves: wave w -> M sub-tiles (w&3) and 4+(w&3), N tile base (w>>2)*16.
// Per K=32 step per wave: 2 A-frags + 4 B-frags feed 8 x
// v_wmma_f32_16x16x32_bf16 (B-frag reused across the two M sub-tiles).
//
// Double-buffered fused LDS block (A at +0, B at +4096 elems; buffer stride
// 8192 elems) toggled with a runtime XOR so the K-loop keeps ONE body and the
// accumulators keep one register assignment (no cross-copies / hazard NOPs).
// Staging uses GLOBAL_LOAD_ASYNC_TO_LDS_B128; s_wait_asynccnt 4 waits only
// for the current tile while the next tile's copies are in flight.
//
// Epilogue (C layout: lane l, vgpr v -> M = v + 8*(l>=16), N = l&15):
//   g0 = kv0 * sigmoid(gate0), g1 = kv1 * sigmoid(gate1)
//   compA[n,d] = sum_r g0[4n+r,d]*ape[r  ,d]     (in-lane reduce over v%4)
//   compB[n,d] = sum_r g1[4n+r,d]*ape[4+r,d]     (feeds comp[n+1])
// ---------------------------------------------------------------------------
__global__ __launch_bounds__(256) void gemm_compress_kernel(
    const bf16_t* __restrict__ xp,      // [T_TOK, DIM] bf16
    const bf16_t* __restrict__ Wp,      // [1024, DIM]  bf16
    const float*  __restrict__ ape,     // [8, HD] fp32
    float* __restrict__ wsA,            // [NGRP, HD] fp32
    float* __restrict__ wsB)            // [NGRP, HD] fp32
{
    // Fused double buffer: [buf][ A: 128x32 (4096 elems) | B: 4x32x32 (4096) ]
    __shared__ __align__(16) bf16_t lds[2 * 8192];      // 32 KB
    const int BUFS = 8192;                              // buffer stride, elements

    const int tid  = threadIdx.x;
    const int lane = tid & 31;
    const int wave = tid >> 5;
    const int mq   = wave & 3;          // M sub-tile quadrant
    const int nb   = (wave >> 2) * 16;  // N tile base within the 32-wide d block

    const int t0 = blockIdx.x * 128;    // token base (multiple of RATIO)
    const int d0 = blockIdx.y * 32;     // d-column base

    // ---- per-thread staging indices (2 A chunks + 2 B chunks per thread) ----
    // A: 128 rows x 32 bf16 = 512 x uint4
    const int    arow0 = tid >> 2,           aseg0 = tid & 3;
    const int    arow1 = (tid + 256) >> 2,   aseg1 = (tid + 256) & 3;
    const size_t aSrc0 = (size_t)(t0 + arow0) * DIM + (size_t)aseg0 * 8;
    const size_t aSrc1 = (size_t)(t0 + arow1) * DIM + (size_t)aseg1 * 8;
    const int    aDst0 = arow0 * 32 + aseg0 * 8;
    const int    aDst1 = arow1 * 32 + aseg1 * 8;
    // B: 4 strips x 32 rows x 32 bf16 = 512 x uint4, at +4096 elements
    const int    bs0   = tid >> 7,           bw0 = (tid >> 2) & 31,           bg0 = tid & 3;
    const int    bs1   = (tid + 256) >> 7,   bw1 = ((tid + 256) >> 2) & 31,   bg1 = (tid + 256) & 3;
    const size_t bSrc0 = (size_t)(bs0 * 256 + d0 + bw0) * DIM + (size_t)bg0 * 8;
    const size_t bSrc1 = (size_t)(bs1 * 256 + d0 + bw1) * DIM + (size_t)bg1 * 8;
    const int    bDst0 = 4096 + (bs0 * 32 + bw0) * 32 + bg0 * 8;
    const int    bDst1 = 4096 + (bs1 * 32 + bw1) * 32 + bg1 * 8;

    // ---- per-wave fragment offsets (LDS element units, within a buffer) ----
    const int fkb   = (lane < 16) ? 0 : 8;    // A frag K sub-block
    const int fkr   = (lane < 16) ? 0 : 8;    // B frag K sub-block
    const int fn    = nb + (lane & 15);
    const int aOffL = (mq * 16 + (lane & 15)) * 32 + fkb;
    const int aOffH = (64 + mq * 16 + (lane & 15)) * 32 + fkb;
    const int bOff  = 4096 + fn * 32 + fkr;

    v8f acc[8] = {};   // acc[2*s + h]: strip s (0=kv0,1=kv1,2=gate0,3=gate1), M-half h

    auto stage = [&](int bufOff, int k0) {
        g2l_b128(xp + aSrc0 + k0, &lds[bufOff + aDst0]);
        g2l_b128(xp + aSrc1 + k0, &lds[bufOff + aDst1]);
        g2l_b128(Wp + bSrc0 + k0, &lds[bufOff + bDst0]);
        g2l_b128(Wp + bSrc1 + k0, &lds[bufOff + bDst1]);
    };

    auto compute = [&](int bufOff) {
        // A fragments (16x32 bf16): chunks K=[fkb,fkb+8) and K=[fkb+16,fkb+24)
        Frag32B fa0, fa1;
        {
            const uint4* p = (const uint4*)(&lds[bufOff + aOffL]);
            fa0.u[0] = p[0]; fa0.u[1] = p[2];
        }
        {
            const uint4* p = (const uint4*)(&lds[bufOff + aOffH]);
            fa1.u[0] = p[0]; fa1.u[1] = p[2];
        }
        // 4 strips: one B fragment (32x16) reused for both M sub-tiles
#pragma unroll
        for (int s = 0; s < 4; ++s) {
            Frag32B fb;
            const uint4* pb = (const uint4*)(&lds[bufOff + s * 1024 + bOff]);
            fb.u[0] = pb[0]; fb.u[1] = pb[2];
            acc[2 * s + 0] = __builtin_amdgcn_wmma_f32_16x16x32_bf16(
                false, fa0.v, false, fb.v, (short)0, acc[2 * s + 0], false, false);
            acc[2 * s + 1] = __builtin_amdgcn_wmma_f32_16x16x32_bf16(
                false, fa1.v, false, fb.v, (short)0, acc[2 * s + 1], false, false);
        }
    };

    int cur = 0;
    stage(cur, 0);
#pragma clang loop unroll(disable)
    for (int k0 = 0; k0 < DIM - 32; k0 += 32) {
        stage(cur ^ BUFS, k0 + 32);    // overlap next-tile copies with compute
        wait_async<4>();               // wait only for the current tile's 4 copies
        __syncthreads();
        compute(cur);
        __syncthreads();
        cur ^= BUFS;
    }
    wait_async<0>();
    __syncthreads();
    compute(cur);                      // last K step (no new staging)

    // ------------------------- epilogue -------------------------
    const int d = d0 + nb + (lane & 15);     // global d column for this lane

    float aA[4], aB[4];
#pragma unroll
    for (int r = 0; r < 4; ++r) {
        aA[r] = ape[r * HD + d];             // ape2[0][r][d]
        aB[r] = ape[(4 + r) * HD + d];       // ape2[1][r][d]
    }

    // token(v,h) = t0 + h*64 + mq*16 + v + 8*(lane>=16); group = token/4, r = v&3
#pragma unroll
    for (int h = 0; h < 2; ++h) {
        const int gbase = blockIdx.x * 32 + h * 16 + mq * 4 + 2 * (lane >> 4);
#pragma unroll
        for (int j = 0; j < 2; ++j) {        // two groups per lane per M-half
            float pA = 0.0f, pB = 0.0f;
#pragma unroll
            for (int r = 0; r < 4; ++r) {
                int   v   = j * 4 + r;
                float g0v = acc[0 + h][v] * (1.0f / (1.0f + __expf(-acc[4 + h][v])));
                float g1v = acc[2 + h][v] * (1.0f / (1.0f + __expf(-acc[6 + h][v])));
                pA += g0v * aA[r];
                pB += g1v * aB[r];
            }
            size_t off = (size_t)(gbase + j) * HD + d;
            wsA[off] = pA;
            wsB[off] = pB;
        }
    }
}

// ---------------------------------------------------------------------------
// Kernel 2: comp[n] = compA[n] + compB[n-1]; RMSNorm; RoPE(first 64, fp32);
//           scatter into out[loc[n]]. One block per position, 256 thr = 1/d.
// ---------------------------------------------------------------------------
__global__ __launch_bounds__(256) void finalize_kernel(
    const float* __restrict__ wsA,
    const float* __restrict__ wsB,
    const float* __restrict__ norm_weight,
    const float* __restrict__ cosT,      // [NGRP, 32]
    const float* __restrict__ sinT,      // [NGRP, 32]
    const int*   __restrict__ loc,
    float* __restrict__ out)             // [POOL, HD]
{
    __shared__ float sh[256];
    __shared__ float sy[256];

    const int n = blockIdx.x;
    const int d = threadIdx.x;

    float comp = wsA[(size_t)n * HD + d];
    if (n > 0) comp += wsB[(size_t)(n - 1) * HD + d];

    sh[d] = comp * comp;
    __syncthreads();
#pragma unroll
    for (int off = 128; off > 0; off >>= 1) {
        if (d < off) sh[d] += sh[d + off];
        __syncthreads();
    }
    const float rms = rsqrtf(sh[0] * (1.0f / (float)HD) + 1e-6f);

    float y = comp * rms * norm_weight[d];
    sy[d] = y;
    __syncthreads();

    float val = y;
    if (d < ROPE_D) {
        const int   p  = d >> 1;
        const float c  = cosT[(size_t)n * (ROPE_D / 2) + p];
        const float s  = sinT[(size_t)n * (ROPE_D / 2) + p];
        const float y1 = sy[2 * p];
        const float y2 = sy[2 * p + 1];
        val = (d & 1) ? (y1 * s + y2 * c) : (y1 * c - y2 * s);
    }

    out[(size_t)loc[n] * HD + d] = val;
}

// ---------------------------------------------------------------------------
// launch
// ---------------------------------------------------------------------------
extern "C" void kernel_launch(void* const* d_in, const int* in_sizes, int n_in,
                              void* d_out, int out_size, void* d_ws, size_t ws_size,
                              hipStream_t stream) {
    const bf16_t* x    = (const bf16_t*)d_in[0];   // [T, DIM] bf16
    const bf16_t* W    = (const bf16_t*)d_in[1];   // [1024, DIM] bf16
    const float*  ape  = (const float*)d_in[2];    // [8, HD]
    const float*  nw   = (const float*)d_in[3];    // [HD]
    const float*  cosT = (const float*)d_in[4];    // [NGRP, 32]
    const float*  sinT = (const float*)d_in[5];    // [NGRP, 32]
    const float*  kvb  = (const float*)d_in[6];    // [POOL, HD]
    const int*    loc  = (const int*)d_in[7];      // [NGRP]
    float*        out  = (float*)d_out;            // [POOL, HD]

    float* wsA = (float*)d_ws;                     // [NGRP, HD] fp32
    float* wsB = wsA + (size_t)NGRP * HD;          // [NGRP, HD] fp32 (16 MB total)

    // 1) copy kv_buffer -> out (out is poisoned; untouched pool rows must be 0)
    pool_copy_kernel<<<(POOL * HD) / 4 / 256, 256, 0, stream>>>(
        (const float4*)kvb, (float4*)out);

    // 2) fused bf16 WMMA GEMM + gating + window-compress partials
    dim3 grid1(T_TOK / 128, HD / 32);   // 256 x 8 blocks
    gemm_compress_kernel<<<grid1, 256, 0, stream>>>(x, W, ape, wsA, wsB);

    // 3) RMSNorm + RoPE + scatter
    finalize_kernel<<<NGRP, 256, 0, stream>>>(wsA, wsB, nw, cosT, sinT, loc, out);
}
